// GoldenMoELayer_53102975648182
// MI455X (gfx1250) — matrix operand order repaired and checked by reading, not compiled, not run on hardware
//
#include <hip/hip_runtime.h>
#include <hip/hip_bf16.h>
#include <math.h>

#define BT    4096
#define DIMD  2048
#define NE    8
#define FF    2048
#define RR    64
#define NLORA 2

typedef __attribute__((ext_vector_type(16))) __bf16 v16bf;
typedef __attribute__((ext_vector_type(8)))  float  v8f;
typedef __attribute__((ext_vector_type(4)))  int    v4i;

#define ASG __attribute__((address_space(1)))
#define ASL __attribute__((address_space(3)))

#if __has_builtin(__builtin_amdgcn_global_load_async_to_lds_b128) && \
    __has_builtin(__builtin_amdgcn_s_wait_asynccnt)
#define USE_ASYNC 1
#else
#define USE_ASYNC 0
#endif

union FragB { uint4 q[2]; v16bf v; };

__device__ __forceinline__ unsigned short f32_bf16(float f) {
  union { float f; unsigned int u; } c; c.f = f;
  unsigned int u = c.u;
  u += 0x7FFFu + ((u >> 16) & 1u);   // round-to-nearest-even
  return (unsigned short)(u >> 16);
}
__device__ __forceinline__ float bf16_f32(unsigned short h) {
  union { unsigned int u; float f; } c; c.u = ((unsigned int)h) << 16;
  return c.f;
}
// pack two f32 -> packed bf16 dword (lo = a, hi = b)
__device__ __forceinline__ unsigned int pack_bf16(float a, float b) {
#if __has_builtin(__builtin_amdgcn_cvt_pk_bf16_f32)
  auto pk = __builtin_amdgcn_cvt_pk_bf16_f32(a, b);
  unsigned int u;
  __builtin_memcpy(&u, &pk, 4);
  return u;
#else
  return (unsigned int)f32_bf16(a) | ((unsigned int)f32_bf16(b) << 16);
#endif
}

__device__ __forceinline__ void wait_async_barrier() {
#if USE_ASYNC
  __builtin_amdgcn_s_wait_asynccnt(0);
#endif
  __syncthreads();
}

// ---------------------------------------------------------------- router
__global__ __launch_bounds__(256) void router_kernel(
    const float* __restrict__ x, const float* __restrict__ Wr,
    const float* __restrict__ temp, float* __restrict__ wts) {
  __shared__ float red[256][NE];
  const int t = blockIdx.x;
  const int tid = threadIdx.x;
  float acc[NE];
  #pragma unroll
  for (int e = 0; e < NE; ++e) acc[e] = 0.f;
  const float* xr = x + (size_t)t * DIMD;
  for (int d = tid; d < DIMD; d += 256) {
    float xv = xr[d];
    const float* wr = Wr + (size_t)d * NE;
    #pragma unroll
    for (int e = 0; e < NE; ++e) acc[e] += xv * wr[e];
  }
  #pragma unroll
  for (int e = 0; e < NE; ++e) red[tid][e] = acc[e];
  __syncthreads();
  for (int s = 128; s > 0; s >>= 1) {
    if (tid < s) {
      #pragma unroll
      for (int e = 0; e < NE; ++e) red[tid][e] += red[tid + s][e];
    }
    __syncthreads();
  }
  if (tid == 0) {
    const float GC = 0.36787944117144233f;   // 1/e
    const float GL = 0.21231792754821910f;   // 0.5 - ln(4/3)
    const float GU = 0.5f;
    const float tp = temp[0];
    float dist[NE], w[NE], fb[NE];
    float wsum = 0.f;
    for (int e = 0; e < NE; ++e) {
      float inh = 1.f / (1.f + __expf(-red[0][e] / tp));
      dist[e] = fabsf(inh - GC);
      bool zone = (inh >= GL) && (inh <= GU);
      w[e] = zone ? __expf(-dist[e] * 10.f) : 0.f;
      wsum += w[e];
    }
    if (wsum < 1e-8f) {
      for (int e = 0; e < NE; ++e) fb[e] = __expf(-dist[e] * (1.f / 0.3f));
      int i1 = 0;
      for (int e = 1; e < NE; ++e) if (fb[e] > fb[i1]) i1 = e;
      int i2 = (i1 == 0) ? 1 : 0;
      for (int e = 0; e < NE; ++e) if (e != i1 && fb[e] > fb[i2]) i2 = e;
      float s = fmaxf(fb[i1] + fb[i2], 1e-8f);
      wsum = 0.f;
      for (int e = 0; e < NE; ++e) { w[e] = (e == i1 || e == i2) ? fb[e] / s : 0.f; wsum += w[e]; }
    }
    wsum = fmaxf(wsum, 1e-8f);
    for (int e = 0; e < NE; ++e) wts[(size_t)t * NE + e] = w[e] / wsum;
  }
}

// ---------------------------------------------------------------- f32 -> bf16
__global__ __launch_bounds__(256) void cvt_bf16_kernel(
    const float* __restrict__ src, unsigned short* __restrict__ dst, int n) {
  int i = (blockIdx.x * 256 + threadIdx.x) * 4;
  if (i + 3 < n) {
    float4 v = *(const float4*)(src + i);
    unsigned int lo = pack_bf16(v.x, v.y);
    unsigned int hi = pack_bf16(v.z, v.w);
    *(uint2*)(dst + i) = make_uint2(lo, hi);
  }
}

// ------------------------------------------------------- LoRA A-projections
__global__ __launch_bounds__(256) void lora_proj_f32(
    const float* __restrict__ X, const float* __restrict__ A,
    unsigned short* __restrict__ out, int K) {
  const int r = threadIdx.x & (RR - 1);
  const int t = blockIdx.x * 4 + (threadIdx.x >> 6);
  const float* xr = X + (size_t)t * K;
  float acc = 0.f;
  #pragma unroll 4
  for (int k = 0; k < K; ++k) acc += xr[k] * A[(size_t)k * RR + r];
  out[(size_t)t * RR + r] = f32_bf16(acc);
}

__global__ __launch_bounds__(256) void lora_proj_bf16(
    const unsigned short* __restrict__ X, const float* __restrict__ A,
    unsigned short* __restrict__ out, int K) {
  const int r = threadIdx.x & (RR - 1);
  const int t = blockIdx.x * 4 + (threadIdx.x >> 6);
  const unsigned short* xr = X + (size_t)t * K;
  float acc = 0.f;
  #pragma unroll 4
  for (int k = 0; k < K; ++k) acc += bf16_f32(xr[k]) * A[(size_t)k * RR + r];
  out[(size_t)t * RR + r] = f32_bf16(acc);
}

// ---------------------------------------------------------------- tiles
#define LDT 40   // LDS row stride in halves (multiple of 8 -> 16B-aligned rows)

// A tile (128 x 32 bf16): straight copy, async global->LDS when available.
// Layout: As[row][k] row stride LDT.
__device__ __forceinline__ void copy_a_tile(
    const unsigned short* __restrict__ src, size_t stride, int t0, int kofs,
    unsigned short* dst, int arow, int aseg) {
  const unsigned short* g = src + (size_t)(t0 + arow) * stride + kofs + aseg;
  unsigned short* l = dst + arow * LDT + aseg;
#if USE_ASYNC
  __builtin_amdgcn_global_load_async_to_lds_b128((ASG v4i*)g, (ASL v4i*)l, 0, 0);
  __builtin_amdgcn_global_load_async_to_lds_b128((ASG v4i*)(g + 8), (ASL v4i*)(l + 8), 0, 0);
#else
  *(uint4*)l = *(const uint4*)g;
  *(uint4*)(l + 8) = *(const uint4*)(g + 8);
#endif
}

// B tile (32 k x 128 n, f32) loaded into regs; stored transposed [n][k] as bf16.
// Each thread: k-pair kp = tid&15 (k = 2kp, 2kp+1), n block fblk = tid>>4 (8 n).
struct Rows8 { float4 a, b; };
struct RowPair { Rows8 r0, r1; };

__device__ __forceinline__ RowPair load_bpair(const float* __restrict__ W,
                                              size_t ld, int k0, int n0,
                                              int kp, int fblk) {
  const float* p = W + (size_t)(k0 + 2 * kp) * ld + n0 + fblk * 8;
  RowPair rp;
  rp.r0.a = ((const float4*)p)[0];      rp.r0.b = ((const float4*)p)[1];
  rp.r1.a = ((const float4*)(p + ld))[0]; rp.r1.b = ((const float4*)(p + ld))[1];
  return rp;
}
__device__ __forceinline__ void store_bpair(unsigned short* B, int kp, int fblk,
                                            const RowPair& rp) {
  float e0[8] = {rp.r0.a.x, rp.r0.a.y, rp.r0.a.z, rp.r0.a.w,
                 rp.r0.b.x, rp.r0.b.y, rp.r0.b.z, rp.r0.b.w};
  float e1[8] = {rp.r1.a.x, rp.r1.a.y, rp.r1.a.z, rp.r1.a.w,
                 rp.r1.b.x, rp.r1.b.y, rp.r1.b.z, rp.r1.b.w};
  #pragma unroll
  for (int j = 0; j < 8; ++j)
    *(unsigned int*)&B[(fblk * 8 + j) * LDT + 2 * kp] = pack_bf16(e0[j], e1[j]);
}

// ---------------------------------------------------------------- fragments
__device__ __forceinline__ void load_afrag(FragB* a, const unsigned short* As,
                                           int wm, int lrow, int hsel) {
  #pragma unroll
  for (int mi = 0; mi < 2; ++mi) {
    const unsigned short* p = &As[(wm * 32 + mi * 16 + lrow) * LDT + hsel * 8];
    a[mi].q[0] = *(const uint4*)p;          // K = hsel*8 .. +7
    a[mi].q[1] = *(const uint4*)(p + 16);   // K = 16 + hsel*8 .. +7
  }
}
__device__ __forceinline__ FragB load_bfrag(const unsigned short* Bs,
                                            int wn, int ni, int lrow, int hsel) {
  const unsigned short* p = &Bs[(wn * 64 + ni * 16 + lrow) * LDT + hsel * 8];
  FragB b;
  b.q[0] = *(const uint4*)p;        // K = hsel*8 .. +7
  b.q[1] = *(const uint4*)(p + 16); // K = 16 + hsel*8 .. +7
  return b;
}

template <bool DUAL>
__device__ __forceinline__ void mma_stage(
    const unsigned short* As, const unsigned short* B0, const unsigned short* B1,
    v8f (*acc0)[4], v8f (*acc1)[4], int wm, int wn, int lrow, int hsel) {
  FragB a[2];
  load_afrag(a, As, wm, lrow, hsel);
  #pragma unroll
  for (int ni = 0; ni < 4; ++ni) {
    FragB b0 = load_bfrag(B0, wn, ni, lrow, hsel);
    #pragma unroll
    for (int mi = 0; mi < 2; ++mi)
      acc0[mi][ni] = __builtin_amdgcn_wmma_f32_16x16x32_bf16(
          false, a[mi].v, false, b0.v, (short)0, acc0[mi][ni], false, false);
    if (DUAL) {
      FragB b1 = load_bfrag(B1, wn, ni, lrow, hsel);
      #pragma unroll
      for (int mi = 0; mi < 2; ++mi)
        acc1[mi][ni] = __builtin_amdgcn_wmma_f32_16x16x32_bf16(
            false, a[mi].v, false, b1.v, (short)0, acc1[mi][ni], false, false);
    }
  }
}

// --------------------------------------------------- gate+up fused WMMA GEMM
__global__ __launch_bounds__(256) void gemm_gateup(
    const unsigned short* __restrict__ xb,
    const float* __restrict__ Wg, const float* __restrict__ Wu,
    const unsigned short* __restrict__ lG, const unsigned short* __restrict__ lU,
    const float* __restrict__ Gb, const float* __restrict__ Ub,
    unsigned short* __restrict__ mid, int use_lora) {
  __shared__ __align__(16) unsigned short As[2][128 * LDT];
  __shared__ __align__(16) unsigned short Bgs[2][128 * LDT];
  __shared__ __align__(16) unsigned short Bus[2][128 * LDT];
  const int tid = threadIdx.x;
  const int t0 = blockIdx.y * 128, n0 = blockIdx.x * 128;
  const int lane = tid & 31, wave = tid >> 5;
  const int wm = wave & 3, wn = wave >> 2;
  const int lrow = lane & 15, hsel = lane >> 4;
  const int arow = tid >> 1, aseg = (tid & 1) * 16;
  const int kp = tid & 15, fblk = tid >> 4;

  v8f accG[2][4], accU[2][4];
  #pragma unroll
  for (int mi = 0; mi < 2; ++mi)
    #pragma unroll
    for (int ni = 0; ni < 4; ++ni) { accG[mi][ni] = {}; accU[mi][ni] = {}; }

  // prologue: stage 0
  copy_a_tile(xb, DIMD, t0, 0, As[0], arow, aseg);
  {
    RowPair bg = load_bpair(Wg, FF, 0, n0, kp, fblk);
    RowPair bu = load_bpair(Wu, FF, 0, n0, kp, fblk);
    store_bpair(Bgs[0], kp, fblk, bg);
    store_bpair(Bus[0], kp, fblk, bu);
  }
  wait_async_barrier();

  int cur = 0;
  for (int k0 = 32; k0 < DIMD; k0 += 32) {
    const int nxt = cur ^ 1;
    copy_a_tile(xb, DIMD, t0, k0, As[nxt], arow, aseg);       // async into nxt
    RowPair nbg = load_bpair(Wg, FF, k0, n0, kp, fblk);
    RowPair nbu = load_bpair(Wu, FF, k0, n0, kp, fblk);
    mma_stage<true>(As[cur], Bgs[cur], Bus[cur], accG, accU, wm, wn, lrow, hsel);
    store_bpair(Bgs[nxt], kp, fblk, nbg);
    store_bpair(Bus[nxt], kp, fblk, nbu);
    wait_async_barrier();
    cur = nxt;
  }
  mma_stage<true>(As[cur], Bgs[cur], Bus[cur], accG, accU, wm, wn, lrow, hsel);

  if (use_lora) {
    for (int kt = 0; kt < RR; kt += 32) {
      // gate LoRA pass: A = x@Ga (bf16 [BT][R]), B = Gb [R][F]
      __syncthreads();
      copy_a_tile(lG, RR, t0, kt, As[0], arow, aseg);
      {
        RowPair tb = load_bpair(Gb, FF, kt, n0, kp, fblk);
        store_bpair(Bgs[0], kp, fblk, tb);
      }
      wait_async_barrier();
      mma_stage<false>(As[0], Bgs[0], nullptr, accG, accG, wm, wn, lrow, hsel);
      // up LoRA pass: A = x@Ua, B = Ub
      __syncthreads();
      copy_a_tile(lU, RR, t0, kt, As[0], arow, aseg);
      {
        RowPair tb = load_bpair(Ub, FF, kt, n0, kp, fblk);
        store_bpair(Bgs[0], kp, fblk, tb);
      }
      wait_async_barrier();
      mma_stage<false>(As[0], Bgs[0], nullptr, accU, accU, wm, wn, lrow, hsel);
    }
  }

  // epilogue: mid = silu(gate) * up, bf16
  #pragma unroll
  for (int mi = 0; mi < 2; ++mi) {
    const int trow = t0 + wm * 32 + mi * 16 + hsel * 8;   // C layout: M=(l/16)*8+r
    #pragma unroll
    for (int ni = 0; ni < 4; ++ni) {
      const int fcol = n0 + wn * 64 + ni * 16 + lrow;     // N = l%16
      v8f g = accG[mi][ni], u = accU[mi][ni];
      #pragma unroll
      for (int r = 0; r < 8; ++r) {
        float gv = g[r];
        float m = (gv / (1.f + __expf(-gv))) * u[r];
        mid[(size_t)(trow + r) * FF + fcol] = f32_bf16(m);
      }
    }
  }
}

// --------------------------------------------------------- down WMMA GEMM
__global__ __launch_bounds__(256) void gemm_down(
    const unsigned short* __restrict__ midb, const float* __restrict__ Wd,
    const unsigned short* __restrict__ lD, const float* __restrict__ Db,
    const float* __restrict__ wts, int expert,
    float* __restrict__ out, int use_lora) {
  __shared__ __align__(16) unsigned short As[2][128 * LDT];
  __shared__ __align__(16) unsigned short Bs[2][128 * LDT];
  __shared__ float wrow[128];
  const int tid = threadIdx.x;
  const int t0 = blockIdx.y * 128, n0 = blockIdx.x * 128;
  const int lane = tid & 31, wave = tid >> 5;
  const int wm = wave & 3, wn = wave >> 2;
  const int lrow = lane & 15, hsel = lane >> 4;
  const int arow = tid >> 1, aseg = (tid & 1) * 16;
  const int kp = tid & 15, fblk = tid >> 4;

  if (tid < 128) wrow[tid] = wts[(size_t)(t0 + tid) * NE + expert];

  v8f acc[2][4];
  #pragma unroll
  for (int mi = 0; mi < 2; ++mi)
    #pragma unroll
    for (int ni = 0; ni < 4; ++ni) acc[mi][ni] = {};

  copy_a_tile(midb, FF, t0, 0, As[0], arow, aseg);
  {
    RowPair b = load_bpair(Wd, DIMD, 0, n0, kp, fblk);
    store_bpair(Bs[0], kp, fblk, b);
  }
  wait_async_barrier();

  int cur = 0;
  for (int k0 = 32; k0 < FF; k0 += 32) {
    const int nxt = cur ^ 1;
    copy_a_tile(midb, FF, t0, k0, As[nxt], arow, aseg);
    RowPair nb = load_bpair(Wd, DIMD, k0, n0, kp, fblk);
    mma_stage<false>(As[cur], Bs[cur], nullptr, acc, acc, wm, wn, lrow, hsel);
    store_bpair(Bs[nxt], kp, fblk, nb);
    wait_async_barrier();
    cur = nxt;
  }
  mma_stage<false>(As[cur], Bs[cur], nullptr, acc, acc, wm, wn, lrow, hsel);

  if (use_lora) {
    for (int kt = 0; kt < RR; kt += 32) {
      __syncthreads();
      copy_a_tile(lD, RR, t0, kt, As[0], arow, aseg);
      {
        RowPair tb = load_bpair(Db, DIMD, kt, n0, kp, fblk);
        store_bpair(Bs[0], kp, fblk, tb);
      }
      wait_async_barrier();
      mma_stage<false>(As[0], Bs[0], nullptr, acc, acc, wm, wn, lrow, hsel);
    }
  }

  #pragma unroll
  for (int mi = 0; mi < 2; ++mi) {
    #pragma unroll
    for (int ni = 0; ni < 4; ++ni) {
      const int dcol = n0 + wn * 64 + ni * 16 + lrow;
      #pragma unroll
      for (int r = 0; r < 8; ++r) {
        const int tl = wm * 32 + mi * 16 + hsel * 8 + r;
        const size_t idx = (size_t)(t0 + tl) * DIMD + dcol;
        out[idx] += wrow[tl] * acc[mi][ni][r];
      }
    }
  }
}

// ---------------------------------------------------------------- launch
extern "C" void kernel_launch(void* const* d_in, const int* in_sizes, int n_in,
                              void* d_out, int out_size, void* d_ws, size_t ws_size,
                              hipStream_t stream) {
  (void)in_sizes; (void)n_in; (void)out_size; (void)ws_size;
  const float* x    = (const float*)d_in[0];
  const float* Wg   = (const float*)d_in[1];
  const float* Wu   = (const float*)d_in[2];
  const float* Wd   = (const float*)d_in[3];
  const float* Ga   = (const float*)d_in[4];
  const float* Gb   = (const float*)d_in[5];
  const float* Ua   = (const float*)d_in[6];
  const float* Ub   = (const float*)d_in[7];
  const float* Da   = (const float*)d_in[8];
  const float* Db   = (const float*)d_in[9];
  const float* Wr   = (const float*)d_in[10];
  const float* temp = (const float*)d_in[11];

  char* ws = (char*)d_ws;
  float*          wts = (float*)ws;                                   // 128 KB
  unsigned short* xb  = (unsigned short*)(ws + (1u << 17));           // 16 MB
  unsigned short* mid = (unsigned short*)(ws + (1u << 17) + 16777216);
  unsigned short* lG  = (unsigned short*)(ws + (1u << 17) + 2 * 16777216);
  unsigned short* lU  = lG + (size_t)BT * RR;
  unsigned short* lD  = lU + (size_t)BT * RR;
  float* out = (float*)d_out;

  (void)hipMemsetAsync(d_out, 0, (size_t)BT * DIMD * sizeof(float), stream);
  router_kernel<<<BT, 256, 0, stream>>>(x, Wr, temp, wts);
  cvt_bf16_kernel<<<(BT * DIMD) / 1024, 256, 0, stream>>>(x, xb, BT * DIMD);

  for (int e = 0; e < NE; ++e) {
    const int ul = (e < NLORA) ? 1 : 0;
    if (ul) {
      lora_proj_f32<<<BT / 4, 256, 0, stream>>>(x, Ga + (size_t)e * DIMD * RR, lG, DIMD);
      lora_proj_f32<<<BT / 4, 256, 0, stream>>>(x, Ua + (size_t)e * DIMD * RR, lU, DIMD);
    }
    gemm_gateup<<<dim3(FF / 128, BT / 128), 256, 0, stream>>>(
        xb, Wg + (size_t)e * DIMD * FF, Wu + (size_t)e * DIMD * FF,
        lG, lU, Gb + (size_t)e * RR * FF, Ub + (size_t)e * RR * FF, mid, ul);
    if (ul)
      lora_proj_bf16<<<BT / 4, 256, 0, stream>>>(mid, Da + (size_t)e * FF * RR, lD, FF);
    gemm_down<<<dim3(DIMD / 128, BT / 128), 256, 0, stream>>>(
        mid, Wd + (size_t)e * FF * DIMD, lD, Db + (size_t)e * RR * DIMD,
        wts, e, out, ul);
  }
}
